// CSOCSSC_V145_52458730553494
// MI455X (gfx1250) — matrix operand order repaired and checked by reading, not compiled, not run on hardware
//
#include <hip/hip_runtime.h>
#include <hip/hip_bf16.h>
#include <math.h>

#define NN   4096
#define DIM  256
#define PPAIR 131072
#define PHB   16384

typedef __bf16 bf16_t;
typedef float  v8f   __attribute__((ext_vector_type(8)));
typedef bf16_t v16bf __attribute__((ext_vector_type(16)));
typedef bf16_t v8bf  __attribute__((ext_vector_type(8)));

// ---- workspace layout ----
// floats:
#define WS_ALPHA 0                // NN floats
#define WS_BOND  (NN)             // 16 block partials
#define WS_CLASH (NN + 16)        // 512
#define WS_HB    (NN + 528)       // 64
#define WS_ELEC  (NN + 592)       // 256
#define WS_SOLV  (NN + 848)       // 256
#define WS_SC    (NN + 1104)      // 256
#define WS_STAGE 8192             // bf16 staging region starts here (32KB offset)
// bf16 region: latRM[NN*DIM], latT[DIM*NN], w1T[DIM*DIM]  (~4.3MB total ws)

__device__ __forceinline__ float wave_reduce(float v) {
#pragma unroll
    for (int off = 16; off > 0; off >>= 1)
        v += __shfl_xor(v, off, 32);
    return v;
}

// deterministic block reduction (256 threads = 8 waves); result valid on thread 0
__device__ __forceinline__ float block_reduce(float v, float* tmp8) {
    v = wave_reduce(v);
    int wave = threadIdx.x >> 5, lane = threadIdx.x & 31;
    if (lane == 0) tmp8[wave] = v;
    __syncthreads();
    float r = 0.0f;
    if (threadIdx.x == 0) {
#pragma unroll
        for (int i = 0; i < 8; ++i) r += tmp8[i];
    }
    __syncthreads();
    return r;
}

// ---------------- one-time bf16 conversion / transpose prep ----------------
__global__ __launch_bounds__(256) void k_prep_latent(const float* __restrict__ latent,
                                                     bf16_t* __restrict__ latRM,
                                                     bf16_t* __restrict__ latT) {
    int idx = blockIdx.x * 256 + threadIdx.x;   // NN*DIM total
    int i = idx >> 8, c = idx & 255;
    float v = latent[idx];
    bf16_t b = (bf16_t)v;
    latRM[idx] = b;                 // row-major bf16 (WMMA A feed)
    latT[c * NN + i] = b;           // col-major bf16, k-contiguous (WMMA B feed)
}

__global__ __launch_bounds__(256) void k_prep_w1(const float* __restrict__ w1,
                                                 bf16_t* __restrict__ w1T) {
    int idx = blockIdx.x * 256 + threadIdx.x;   // DIM*DIM total
    int k = idx >> 8, c = idx & 255;
    w1T[c * DIM + k] = (bf16_t)w1[idx];         // [col][k]
}

// ---------------- bond / clash / hb partial kernels ----------------
__global__ __launch_bounds__(256) void k_bond(const float* __restrict__ coords,
                                              float* __restrict__ ws) {
    __shared__ float tmp8[8];
    int i = blockIdx.x * 256 + threadIdx.x;
    float s = 0.0f;
    if (i < NN - 1) {
        float dx = coords[3*(i+1)+0] - coords[3*i+0];
        float dy = coords[3*(i+1)+1] - coords[3*i+1];
        float dz = coords[3*(i+1)+2] - coords[3*i+2];
        float d  = sqrtf(dx*dx + dy*dy + dz*dz + 1e-12f) - 3.8f;
        s = d * d;
    }
    float r = block_reduce(s, tmp8);
    if (threadIdx.x == 0) ws[WS_BOND + blockIdx.x] = r;
}

__global__ __launch_bounds__(256) void k_clash(const float* __restrict__ coords,
                                               const int* __restrict__ pairs,
                                               float* __restrict__ ws) {
    __shared__ float tmp8[8];
    int p = blockIdx.x * 256 + threadIdx.x;
    int a = pairs[2*p], b = pairs[2*p+1];
    float dx = coords[3*a+0] - coords[3*b+0];
    float dy = coords[3*a+1] - coords[3*b+1];
    float dz = coords[3*a+2] - coords[3*b+2];
    float d  = sqrtf(dx*dx + dy*dy + dz*dz + 1e-12f);
    float t  = fmaxf(3.2f - d, 0.0f);
    float r = block_reduce(t * t, tmp8);
    if (threadIdx.x == 0) ws[WS_CLASH + blockIdx.x] = r;
}

__global__ __launch_bounds__(256) void k_hb(const float* __restrict__ coords,
                                            const int* __restrict__ hbp,
                                            float* __restrict__ ws) {
    __shared__ float tmp8[8];
    int p = blockIdx.x * 256 + threadIdx.x;
    int a = hbp[2*p], b = hbp[2*p+1];
    float dx = coords[3*a+0] - coords[3*b+0];
    float dy = coords[3*a+1] - coords[3*b+1];
    float dz = coords[3*a+2] - coords[3*b+2];
    float d  = sqrtf(dx*dx + dy*dy + dz*dz + 1e-12f) - 2.9f;
    float r = block_reduce(d * d, tmp8);
    if (threadIdx.x == 0) ws[WS_HB + blockIdx.x] = r;
}

// ---------------- alpha MLP: GELU(latent@w1+b1)@w2+b2 -> sigmoid ----------------
// A and B fragments come straight from prepped bf16 global arrays: no LDS, no barriers in k-loop.
__global__ __launch_bounds__(256) void k_alpha(const bf16_t* __restrict__ latRM,
                                               const bf16_t* __restrict__ w1T,
                                               const float* __restrict__ b1,
                                               const float* __restrict__ w2,
                                               const float* __restrict__ b2,
                                               float* __restrict__ alpha) {
    __shared__ float waveRow[8][16];

    const int t = threadIdx.x;
    const int wave = t >> 5, lane = t & 31;
    const int i0 = blockIdx.x * 16;

    const int arowOff = (i0 + (lane & 15)) * DIM + (lane >> 4) * 16;  // A: row, kb
    const int khalf   = (lane >> 4) * 16;
    const int bcol0   = (wave * 32 + (lane & 15)) * DIM;              // B: col*DIM
    const int bcol1   = bcol0 + 16 * DIM;

    v8f c0 = {}, c1 = {};
#pragma unroll
    for (int k0 = 0; k0 < DIM; k0 += 32) {
        v8bf alo = *(const v8bf*)&latRM[arowOff + k0];
        v8bf ahi = *(const v8bf*)&latRM[arowOff + k0 + 8];
        v16bf a = __builtin_shufflevector(alo, ahi, 0,1,2,3,4,5,6,7,8,9,10,11,12,13,14,15);
        v16bf bf0 = *(const v16bf*)&w1T[bcol0 + k0 + khalf];
        v16bf bf1 = *(const v16bf*)&w1T[bcol1 + k0 + khalf];
        c0 = __builtin_amdgcn_wmma_f32_16x16x32_bf16(false, a, false, bf0, (short)0, c0, false, false);
        c1 = __builtin_amdgcn_wmma_f32_16x16x32_bf16(false, a, false, bf1, (short)0, c1, false, false);
    }

    // epilogue: bias + exact GELU + dot w2, deterministic wave reduce
    const int col0 = wave * 32 + (lane & 15);
    const int col1 = col0 + 16;
    const float w2a = w2[col0], w2b = w2[col1];
    const float b1a = b1[col0], b1b = b1[col1];
#pragma unroll
    for (int v = 0; v < 8; ++v) {
        float h0 = c0[v] + b1a;
        float h1 = c1[v] + b1b;
        float g0 = 0.5f * h0 * (1.0f + erff(h0 * 0.70710678f));
        float g1 = 0.5f * h1 * (1.0f + erff(h1 * 0.70710678f));
        float p  = g0 * w2a + g1 * w2b;
#pragma unroll
        for (int off = 1; off <= 8; off <<= 1) p += __shfl_xor(p, off, 32);
        if ((lane & 15) == 0) waveRow[wave][v + 8 * (lane >> 4)] = p;
    }
    __syncthreads();
    if (t < 16) {
        float s = 0.0f;
#pragma unroll
        for (int w = 0; w < 8; ++w) s += waveRow[w][t];
        s += b2[0];
        alpha[i0 + t] = 0.5f + 2.5f / (1.0f + __expf(-s));
    }
}

// ---------------- fused N^2 pass: K tiles (LDS, double-buffered) -> WMMA K@latT ----------------
// Transcendental folding: u = exp(-De/60) gives exp(-De/12)=u^5, exp(-De/10)=u^6, exp(-d/6)~u^10,
// cutting TRANS ops per element-row from 5 to 3 (log, pow-exp, u) -- the VALU roofline limiter.
__global__ __launch_bounds__(256) void k_main(const float* __restrict__ coords,
                                              const bf16_t* __restrict__ latT,
                                              const float* __restrict__ charges,
                                              const float* __restrict__ hydro,
                                              const float* __restrict__ alpha,
                                              float* __restrict__ out,
                                              float* __restrict__ ws) {
    __shared__ bf16_t ldsK[2][16 * 32] __attribute__((aligned(32)));
    __shared__ float  ldsInv[16];
    __shared__ float  tmp8[8];

    const int t = threadIdx.x;
    const int wave = t >> 5, lane = t & 31;
    const int i0 = blockIdx.x * 16;
    const int cc = t & 31;           // column within K tile (all j-steps)

    // this thread owns K rows (wave) and (wave+8) of the 16-row tile
    const int ia = i0 + wave, ib = i0 + wave + 8;
    const float ax = coords[3*ia+0], ay = coords[3*ia+1], az = coords[3*ia+2];
    const float bx = coords[3*ib+0], by = coords[3*ib+1], bz = coords[3*ib+2];
    const float alA = alpha[ia], alB = alpha[ib];
    const float qA = charges[ia], qB = charges[ib];
    const float hA = hydro[ia],  hB = hydro[ib];

    // fragment addressing (constant across loop)
    const int arow = lane & 15, kb = (lane >> 4) * 16;
    const int khalf = (lane >> 4) * 16;
    const int bcol0 = (wave * 32 + (lane & 15)) * NN;
    const int bcol1 = bcol0 + 16 * NN;

    float rsA = 0.0f, rsB = 0.0f;
    float solv = 0.0f, elec = 0.0f, sc = 0.0f;
    v8f c0 = {}, c1 = {};

    int buf = 0;
    for (int j0 = 0; j0 < NN; j0 += 32, buf ^= 1) {
        const int j = j0 + cc;
        const float cx = coords[3*j+0], cy = coords[3*j+1], cz = coords[3*j+2];
        const float alJ = alpha[j], qJ = charges[j];

        { // row ia
            float dx = ax - cx, dy = ay - cy, dz = az - cz;
            float d  = sqrtf(dx*dx + dy*dy + dz*dz + 1e-12f);
            float De = d + 1e-6f;
            float u  = __expf(De * (-1.0f/60.0f));
            float u2 = u * u, u4 = u2 * u2;
            float u5 = u4 * u, u6 = u4 * u2, u10 = u5 * u5;
            solv += u10;                               // ~exp(-d/6)
            if (d < 8.0f) sc += -0.1f * hA;
            if (j > ia)   elec += qA * qJ * u6 * __builtin_amdgcn_rcpf(80.0f * De);
            float ae = 0.5f * (alA + alJ);
            float kv = __expf(-ae * __logf(De));
            kv = fminf(fmaxf(kv, 1e-8f), 1000.0f) * u5; // * exp(-De/12)
            if (j == ia) kv = 0.0f;
            rsA += kv;
            ldsK[buf][wave * 32 + cc] = (bf16_t)kv;
        }
        { // row ib
            float dx = bx - cx, dy = by - cy, dz = bz - cz;
            float d  = sqrtf(dx*dx + dy*dy + dz*dz + 1e-12f);
            float De = d + 1e-6f;
            float u  = __expf(De * (-1.0f/60.0f));
            float u2 = u * u, u4 = u2 * u2;
            float u5 = u4 * u, u6 = u4 * u2, u10 = u5 * u5;
            solv += u10;
            if (d < 8.0f) sc += -0.1f * hB;
            if (j > ib)   elec += qB * qJ * u6 * __builtin_amdgcn_rcpf(80.0f * De);
            float ae = 0.5f * (alB + alJ);
            float kv = __expf(-ae * __logf(De));
            kv = fminf(fmaxf(kv, 1e-8f), 1000.0f) * u5;
            if (j == ib) kv = 0.0f;
            rsB += kv;
            ldsK[buf][(wave + 8) * 32 + cc] = (bf16_t)kv;
        }
        __syncthreads();   // single barrier per step: buffers alternate

        // A = K tile (16x32 bf16) from LDS; B = latT columns straight from global (L2-resident)
        v8bf alo = *(const v8bf*)&ldsK[buf][arow * 32 + kb];
        v8bf ahi = *(const v8bf*)&ldsK[buf][arow * 32 + kb + 8];
        v16bf a = __builtin_shufflevector(alo, ahi, 0,1,2,3,4,5,6,7,8,9,10,11,12,13,14,15);
        v16bf bf0 = *(const v16bf*)&latT[bcol0 + j0 + khalf];
        v16bf bf1 = *(const v16bf*)&latT[bcol1 + j0 + khalf];
        c0 = __builtin_amdgcn_wmma_f32_16x16x32_bf16(false, a, false, bf0, (short)0, c0, false, false);
        c1 = __builtin_amdgcn_wmma_f32_16x16x32_bf16(false, a, false, bf1, (short)0, c1, false, false);
    }

    // deterministic row-sum reduce (each wave's 32 lanes covered all columns of rows wave, wave+8)
    float sA = wave_reduce(rsA);
    float sB = wave_reduce(rsB);
    if (lane == 0) {
        ldsInv[wave]     = 1.0f / (sA + 1e-8f);
        ldsInv[wave + 8] = 1.0f / (sB + 1e-8f);
    }
    __syncthreads();

    // write normalized diffusion output
#pragma unroll
    for (int v = 0; v < 8; ++v) {
        int m = v + 8 * (lane >> 4);
        float inv = ldsInv[m];
        int col = wave * 32 + (lane & 15);
        out[(i0 + m) * DIM + col]      = c0[v] * inv;
        out[(i0 + m) * DIM + col + 16] = c1[v] * inv;
    }

    // deterministic block partials for scalar energies
    float rE = block_reduce(elec, tmp8);
    float rS = block_reduce(solv, tmp8);
    float rC = block_reduce(sc,   tmp8);
    if (t == 0) {
        ws[WS_ELEC + blockIdx.x] = rE;
        ws[WS_SOLV + blockIdx.x] = rS;
        ws[WS_SC   + blockIdx.x] = rC;
    }
}

// ---------------- finalize: deterministic sequential sums ----------------
__global__ void k_final(const float* __restrict__ ws, float* __restrict__ out) {
    if (threadIdx.x != 0 || blockIdx.x != 0) return;
    float bond = 0.0f, clash = 0.0f, hb = 0.0f, elec = 0.0f, solv = 0.0f, sc = 0.0f;
    for (int i = 0; i < 16; ++i)  bond  += ws[WS_BOND + i];
    for (int i = 0; i < 512; ++i) clash += ws[WS_CLASH + i];
    for (int i = 0; i < 64; ++i)  hb    += ws[WS_HB + i];
    for (int i = 0; i < 256; ++i) elec  += ws[WS_ELEC + i];
    for (int i = 0; i < 256; ++i) solv  += ws[WS_SOLV + i];
    for (int i = 0; i < 256; ++i) sc    += ws[WS_SC + i];
    float e = 20.0f * bond / (float)(NN - 1)
            + 50.0f * clash / (float)PPAIR
            + 3.0f  * elec
            + 8.0f  * hb / (float)PHB
            + solv / ((float)NN * (float)NN)
            + sc / (float)NN;
    out[0] = e;
}

extern "C" void kernel_launch(void* const* d_in, const int* in_sizes, int n_in,
                              void* d_out, int out_size, void* d_ws, size_t ws_size,
                              hipStream_t stream) {
    const float* coords  = (const float*)d_in[0];
    const float* latent  = (const float*)d_in[1];
    const float* charges = (const float*)d_in[2];
    const float* hydro   = (const float*)d_in[3];
    const float* w1      = (const float*)d_in[4];
    const float* b1      = (const float*)d_in[5];
    const float* w2      = (const float*)d_in[6];
    const float* b2      = (const float*)d_in[7];
    const int*   pairs   = (const int*)d_in[8];
    const int*   hbp     = (const int*)d_in[9];

    float* out = (float*)d_out;         // [0] = energy, [1..] = out[N,DIM]
    float* ws  = (float*)d_ws;
    float* alpha = ws + WS_ALPHA;
    bf16_t* latRM = (bf16_t*)(ws + WS_STAGE);
    bf16_t* latT  = latRM + NN * DIM;
    bf16_t* w1T   = latT + NN * DIM;

    hipLaunchKernelGGL(k_prep_latent, dim3(NN * DIM / 256), dim3(256), 0, stream,
                       latent, latRM, latT);
    hipLaunchKernelGGL(k_prep_w1, dim3(DIM * DIM / 256), dim3(256), 0, stream, w1, w1T);
    hipLaunchKernelGGL(k_alpha, dim3(NN / 16), dim3(256), 0, stream,
                       latRM, w1T, b1, w2, b2, alpha);
    hipLaunchKernelGGL(k_bond, dim3(16), dim3(256), 0, stream, coords, ws);
    hipLaunchKernelGGL(k_clash, dim3(PPAIR / 256), dim3(256), 0, stream, coords, pairs, ws);
    hipLaunchKernelGGL(k_hb, dim3(PHB / 256), dim3(256), 0, stream, coords, hbp, ws);
    hipLaunchKernelGGL(k_main, dim3(NN / 16), dim3(256), 0, stream,
                       coords, latT, charges, hydro, alpha, out + 1, ws);
    hipLaunchKernelGGL(k_final, dim3(1), dim3(1), 0, stream, ws, out);
}
